// CrossAttention_86406152061077
// MI455X (gfx1250) — matrix operand order repaired and checked by reading, not compiled, not run on hardware
//
#include <hip/hip_runtime.h>
#include <hip/hip_bf16.h>
#include <math.h>

// ---------------------------------------------------------------------------
// CrossAttention for MI455X (gfx1250): bf16 WMMA everywhere, f32 accumulate.
// B=2, N=KV=2048, C=1024, H=16, hd=64.
//  - gemm_bf16: 128x128 block tile, 8 waves of 32x64, K-step 32, prefetch.
//  - rope_permute: f32 [B,N,C] -> bf16 [B,H,N,64] (and transposed V copy).
//  - attn: flash attention; Q/K/V tiles staged via GLOBAL_LOAD_ASYNC_TO_LDS.
// ---------------------------------------------------------------------------

typedef __attribute__((ext_vector_type(16))) __bf16        v16bf;
typedef __attribute__((ext_vector_type(8)))  float         v8f;
typedef __attribute__((ext_vector_type(8)))  unsigned int  v8u;

#define CA_B   2
#define CA_N   2048
#define CA_KV  2048
#define CA_C   1024
#define CA_H   16
#define CA_HD  64

__device__ __forceinline__ unsigned short f2bf(float f) {
    unsigned int u = __builtin_bit_cast(unsigned int, f);
    u += 0x7FFFu + ((u >> 16) & 1u);          // round-to-nearest-even
    return (unsigned short)(u >> 16);
}

// A-fragment (16x32 bf16, MxK): lane holds row m = lane&15; element pairs are
// K = {h*8+2j} for j<4 and {16+h*8+2(j-4)} for j>=4, h = lane>>4.
__device__ __forceinline__ v16bf load_afrag(const unsigned short* rowbase,
                                            int lane, int koff) {
    const int h = lane >> 4;
    v8u a;
#pragma unroll
    for (int j = 0; j < 8; ++j) {
        const int k = (j < 4) ? (h * 8 + 2 * j) : (16 + h * 8 + 2 * (j - 4));
        a[j] = *(const unsigned int*)(rowbase + koff + k);
    }
    return __builtin_bit_cast(v16bf, a);
}

// B-fragment (32x16 bf16, KxN): lane holds column n = lane&15; element pairs
// K = {h*8+2j} for j<4 and {16+h*8+2(j-4)} for j>=4, h = lane>>4.
__device__ __forceinline__ v16bf load_bfrag(const unsigned short* colbase,
                                            int lane, int koff) {
    const int h = lane >> 4;
    v8u b;
#pragma unroll
    for (int j = 0; j < 8; ++j) {
        const int k = (j < 4) ? (h * 8 + 2 * j) : (16 + h * 8 + 2 * (j - 4));
        b[j] = *(const unsigned int*)(colbase + koff + k);
    }
    return __builtin_bit_cast(v16bf, b);
}

__device__ __forceinline__ v8f wmma_bf16(v16bf a, v16bf b, v8f c) {
    return __builtin_amdgcn_wmma_f32_16x16x32_bf16(
        false, a, false, b, (short)0, c, false, false);
}

__device__ __forceinline__ v8f v8f_zero() {
    v8f z = {0.f, 0.f, 0.f, 0.f, 0.f, 0.f, 0.f, 0.f};
    return z;
}

// CDNA5 async copy: global -> LDS, 16 bytes per lane, tracked by ASYNCcnt.
__device__ __forceinline__ void async_g2lds_b128(unsigned int lds_off,
                                                 unsigned long long gaddr) {
    asm volatile("global_load_async_to_lds_b128 %0, %1, off"
                 :: "v"(lds_off), "v"(gaddr) : "memory");
}
__device__ __forceinline__ void wait_asynccnt0() {
    asm volatile("s_wait_asynccnt 0x0" ::: "memory");
}

// ---------------------------------------------------------------------------
// GEMM: C[M,N] (f32) = A[M,K] (f32 -> bf16) * B[K,N] (f32 -> bf16)
// Block tile 128x128, 256 threads (8 waves), wave tile 32x64, K-step 32.
// ---------------------------------------------------------------------------
__global__ __launch_bounds__(256)
void gemm_bf16_kernel(const float* __restrict__ A, const float* __restrict__ Bm,
                      float* __restrict__ C, int M, int N, int K) {
    __shared__ unsigned short sA[128 * 32];   // [row][k]   8 KB
    __shared__ unsigned short sBt[128 * 32];  // [col][k]   8 KB (transposed)

    const int tid  = threadIdx.x;
    const int wave = tid >> 5;
    const int lane = tid & 31;

    const int row0 = blockIdx.y * 128;
    const int col0 = blockIdx.x * 128;
    const int wR   = (wave >> 1) * 32;        // wave row offset (4 rows of waves)
    const int wC   = (wave & 1) * 64;         // wave col offset (2 cols of waves)

    v8f acc[2][4];
#pragma unroll
    for (int i = 0; i < 2; ++i)
#pragma unroll
        for (int j = 0; j < 4; ++j) acc[i][j] = v8f_zero();

    for (int k0 = 0; k0 < K; k0 += 32) {
        // prefetch next k-tile into L2 (global_prefetch_b8)
        if (k0 + 32 < K) {
            const int r = tid >> 1, halfsel = tid & 1;
            __builtin_prefetch(&A[(size_t)(row0 + r) * K + k0 + 32 + halfsel * 16],
                               0, 3);
            const int kk = tid >> 3, n8 = (tid & 7) * 16;
            __builtin_prefetch(&Bm[(size_t)(k0 + 32 + kk) * N + col0 + n8], 0, 3);
        }
        // stage A tile 128x32 (16 elems/thread)
#pragma unroll
        for (int i = 0; i < 16; ++i) {
            const int idx = tid + i * 256;
            const int r = idx >> 5, c = idx & 31;
            sA[idx] = f2bf(A[(size_t)(row0 + r) * K + k0 + c]);
        }
        // stage B tile 32x128 transposed (16 elems/thread)
#pragma unroll
        for (int i = 0; i < 16; ++i) {
            const int idx = tid + i * 256;
            const int kk = idx >> 7, n = idx & 127;
            sBt[n * 32 + kk] = f2bf(Bm[(size_t)(k0 + kk) * N + col0 + n]);
        }
        __syncthreads();

        v16bf af[2], bf[4];
#pragma unroll
        for (int mi = 0; mi < 2; ++mi)
            af[mi] = load_afrag(&sA[(wR + mi * 16 + (lane & 15)) * 32], lane, 0);
#pragma unroll
        for (int ni = 0; ni < 4; ++ni)
            bf[ni] = load_bfrag(&sBt[(wC + ni * 16 + (lane & 15)) * 32], lane, 0);
#pragma unroll
        for (int mi = 0; mi < 2; ++mi)
#pragma unroll
            for (int ni = 0; ni < 4; ++ni)
                acc[mi][ni] = wmma_bf16(af[mi], bf[ni], acc[mi][ni]);
        __syncthreads();
    }

    // C/D layout: row = (lane>>4)*8 + r, col = lane & 15
#pragma unroll
    for (int mi = 0; mi < 2; ++mi)
#pragma unroll
        for (int ni = 0; ni < 4; ++ni)
#pragma unroll
            for (int r = 0; r < 8; ++r) {
                const int row = row0 + wR + mi * 16 + ((lane >> 4) << 3) + r;
                const int col = col0 + wC + ni * 16 + (lane & 15);
                C[(size_t)row * N + col] = acc[mi][ni][r];
            }
}

// ---------------------------------------------------------------------------
// RoPE + permute to bf16: src [B,N,C] f32 -> dst bf16.
//   transposeV == 0: dst [B,H,N,64], optional rotary + scale
//   transposeV == 1: dst [B,H,64,N]  (V^T copy, no rotary)
// 256 threads handle 4 (b,h,l) rows of 64.
// ---------------------------------------------------------------------------
__global__ __launch_bounds__(256)
void rope_permute_kernel(const float* __restrict__ src,
                         const float* __restrict__ pos,
                         unsigned short* __restrict__ dst,
                         int doRope, float scale, int transposeV) {
    const int rid = blockIdx.x * 4 + (threadIdx.x >> 6);   // (b*H + h)*N + l
    const int d   = threadIdx.x & 63;
    const int b   = rid >> 15;                  // / (H*N)
    const int rem = rid & 32767;
    const int h   = rem >> 11;                  // / N
    const int l   = rem & 2047;

    const size_t sidx = (size_t)(b * CA_N + l) * CA_C + h * CA_HD + d;
    float x = src[sidx];
    float out = x;
    if (doRope) {
        const float other = (d < 32) ? -src[sidx + 32] : src[sidx - 32];
        const float pv = pos[l * CA_HD + d];
        out = x * __cosf(pv) + other * __sinf(pv);
    }
    out *= scale;
    const int bh = rid >> 11;                   // b*H + h
    const size_t didx = transposeV
        ? ((size_t)bh * CA_HD + d) * CA_N + l   // [B,H,64,N]
        : (size_t)rid * CA_HD + d;              // [B,H,N,64]
    dst[didx] = f2bf(out);
}

// ---------------------------------------------------------------------------
// Flash attention: grid = (N/64 q-tiles, B*H). 128 threads = 4 waves; each
// wave owns 16 query rows. bf16 WMMA matmuls, f32 online softmax.
// Qb/Kb: [B,H,L,64] bf16.  Vt: [B,H,64,KV] bf16.  O -> [B,N,C] f32.
// Tiles staged with GLOBAL_LOAD_ASYNC_TO_LDS_B128 (ASYNCcnt).
// ---------------------------------------------------------------------------
__global__ __launch_bounds__(128)
void attn_kernel(const unsigned short* __restrict__ Qb,
                 const unsigned short* __restrict__ Kb,
                 const unsigned short* __restrict__ Vt,
                 const int* __restrict__ mask, float* __restrict__ O) {
    __shared__ unsigned short sQ[64 * 64];      // [qrow][d]   8 KB
    __shared__ unsigned short sK[64 * 64];      // [key][d]    8 KB
    __shared__ unsigned short sVt[64 * 64];     // [d][key]    8 KB
    __shared__ unsigned short sP[4][16 * 64];   // per-wave P  8 KB

    const int tid  = threadIdx.x;
    const int wave = tid >> 5;
    const int lane = tid & 31;
    const int qb   = blockIdx.x;                // q tile (64 rows)
    const int bh   = blockIdx.y;                // b*H + h
    const int b    = bh >> 4;
    const int h    = bh & 15;

    const unsigned int ldsQ  = (unsigned int)(uintptr_t)&sQ[0];
    const unsigned int ldsK  = (unsigned int)(uintptr_t)&sK[0];
    const unsigned int ldsVt = (unsigned int)(uintptr_t)&sVt[0];

    // async-stage the (contiguous) 8 KB Q tile: 4 x 16B per thread
    {
        const unsigned long long qbase =
            (unsigned long long)(uintptr_t)(Qb + ((size_t)bh * CA_N + qb * 64) * CA_HD);
#pragma unroll
        for (int i = 0; i < 4; ++i) {
            const unsigned int o = (unsigned int)(tid + i * 128) * 16u;
            async_g2lds_b128(ldsQ + o, qbase + o);
        }
        wait_asynccnt0();
    }
    __syncthreads();

    // Q A-fragments for this wave (rows wave*16..+15), K split d=[0,32),[32,64)
    v16bf aq[2];
#pragma unroll
    for (int s = 0; s < 2; ++s)
        aq[s] = load_afrag(&sQ[(wave * 16 + (lane & 15)) * 64], lane, s * 32);

    v8f oacc[4];
#pragma unroll
    for (int i = 0; i < 4; ++i) oacc[i] = v8f_zero();
    float mrun[8], lrun[8];
#pragma unroll
    for (int r = 0; r < 8; ++r) { mrun[r] = -3.0e38f; lrun[r] = 0.f; }

    for (int j0 = 0; j0 < CA_KV; j0 += 64) {
        __syncthreads();
        // async-stage K chunk (contiguous 8 KB) and V^T chunk (64 rows x 128 B)
        {
            const unsigned long long kbase =
                (unsigned long long)(uintptr_t)(Kb + ((size_t)bh * CA_N + j0) * CA_HD);
#pragma unroll
            for (int i = 0; i < 4; ++i) {
                const unsigned int o = (unsigned int)(tid + i * 128) * 16u;
                async_g2lds_b128(ldsK + o, kbase + o);
            }
            const unsigned long long vbase =
                (unsigned long long)(uintptr_t)(Vt + (size_t)bh * CA_HD * CA_KV + j0);
#pragma unroll
            for (int i = 0; i < 4; ++i) {
                const int idx = tid + i * 128;          // 0..511
                const int drow = idx >> 3, seg = idx & 7;
                async_g2lds_b128(ldsVt + (unsigned int)(drow * 128 + seg * 16),
                                 vbase + (unsigned long long)drow * (CA_KV * 2)
                                       + (unsigned long long)seg * 16);
            }
            wait_asynccnt0();
        }
        __syncthreads();

        // S = Q * K^T  (16 rows x 64 keys per wave)
        v8f sacc[4];
#pragma unroll
        for (int nj = 0; nj < 4; ++nj) {
            sacc[nj] = v8f_zero();
#pragma unroll
            for (int s = 0; s < 2; ++s)
                sacc[nj] = wmma_bf16(
                    aq[s],
                    load_bfrag(&sK[(nj * 16 + (lane & 15)) * 64], lane, s * 32),
                    sacc[nj]);
        }
        // mask[b, j]
#pragma unroll
        for (int nj = 0; nj < 4; ++nj) {
            const int mv = mask[b * CA_KV + j0 + nj * 16 + (lane & 15)];
            if (!mv) {
#pragma unroll
                for (int r = 0; r < 8; ++r) sacc[nj][r] = -3.0e38f;
            }
        }

        // online softmax per row (16 lanes share a row; cols per lane distinct)
#pragma unroll
        for (int r = 0; r < 8; ++r) {
            float vmax = fmaxf(fmaxf(sacc[0][r], sacc[1][r]),
                               fmaxf(sacc[2][r], sacc[3][r]));
#pragma unroll
            for (int m = 1; m < 16; m <<= 1)
                vmax = fmaxf(vmax, __shfl_xor(vmax, m, 32));
            const float mnew = fmaxf(mrun[r], vmax);
            const float corr = __expf(mrun[r] - mnew);
            float p[4], psum = 0.f;
#pragma unroll
            for (int nj = 0; nj < 4; ++nj) {
                p[nj] = __expf(sacc[nj][r] - mnew);
                psum += p[nj];
            }
#pragma unroll
            for (int m = 1; m < 16; m <<= 1) psum += __shfl_xor(psum, m, 32);
            lrun[r] = lrun[r] * corr + psum;
            mrun[r] = mnew;
#pragma unroll
            for (int nd = 0; nd < 4; ++nd) oacc[nd][r] *= corr;
            // spill P (C-layout) to wave-private LDS for re-read in A-layout
            const int prow = ((lane >> 4) << 3) + r;
#pragma unroll
            for (int nj = 0; nj < 4; ++nj)
                sP[wave][prow * 64 + nj * 16 + (lane & 15)] = f2bf(p[nj]);
        }

        // O += P * V
        v16bf pf[2];
#pragma unroll
        for (int s = 0; s < 2; ++s)
            pf[s] = load_afrag(&sP[wave][(lane & 15) * 64], lane, s * 32);
#pragma unroll
        for (int nd = 0; nd < 4; ++nd)
#pragma unroll
            for (int s = 0; s < 2; ++s)
                oacc[nd] = wmma_bf16(
                    pf[s],
                    load_bfrag(&sVt[(nd * 16 + (lane & 15)) * 64], lane, s * 32),
                    oacc[nd]);
    }

    // normalize and write O in [B,N,C] layout (== transpose(0,2,1,3).reshape)
#pragma unroll
    for (int r = 0; r < 8; ++r) {
        const float inv = (lrun[r] > 0.f) ? (1.0f / lrun[r]) : 0.f;
        const int q = qb * 64 + wave * 16 + ((lane >> 4) << 3) + r;
#pragma unroll
        for (int nd = 0; nd < 4; ++nd) {
            const int col = h * CA_HD + nd * 16 + (lane & 15);
            O[((size_t)b * CA_N + q) * CA_C + col] = oacc[nd][r] * inv;
        }
    }
}

// ---------------------------------------------------------------------------
extern "C" void kernel_launch(void* const* d_in, const int* in_sizes, int n_in,
                              void* d_out, int out_size, void* d_ws, size_t ws_size,
                              hipStream_t stream) {
    const float* q     = (const float*)d_in[0];
    const float* k     = (const float*)d_in[1];
    const float* v     = (const float*)d_in[2];
    const int*   mask  = (const int*)d_in[3];
    const float* pos   = (const float*)d_in[4];
    const float* qW    = (const float*)d_in[5];
    const float* kW    = (const float*)d_in[6];
    const float* vW    = (const float*)d_in[7];
    const float* projW = (const float*)d_in[8];
    float*       out   = (float*)d_out;

    const size_t per = (size_t)CA_B * CA_N * CA_C;     // 4 Mi elements
    float* ws = (float*)d_ws;
    float* Qp = ws + 0 * per;                          // f32 projections
    float* Kp = ws + 1 * per;
    float* Vp = ws + 2 * per;
    unsigned short* Qb  = (unsigned short*)(ws + 3 * per);   // bf16 rotated
    unsigned short* Kb  = Qb + per;
    unsigned short* Vbt = Kb + per;                          // bf16 V^T
    float* Ob = Qp;                                    // reuse after RoPE

    const int M = CA_B * CA_N;                         // 4096
    dim3 gGrid(CA_C / 128, M / 128);                   // (8, 32)

    gemm_bf16_kernel<<<gGrid, 256, 0, stream>>>(q, qW, Qp, M, CA_C, CA_C);
    gemm_bf16_kernel<<<gGrid, 256, 0, stream>>>(k, kW, Kp, M, CA_C, CA_C);
    gemm_bf16_kernel<<<gGrid, 256, 0, stream>>>(v, vW, Vp, M, CA_C, CA_C);

    const int ropeBlocks = (CA_B * CA_H * CA_N) / 4;   // 16384
    const float scale = 0.125f;                        // hd^-0.5
    rope_permute_kernel<<<ropeBlocks, 256, 0, stream>>>(Qp, pos, Qb, 1, scale, 0);
    rope_permute_kernel<<<ropeBlocks, 256, 0, stream>>>(Kp, pos, Kb, 1, 1.0f, 0);
    rope_permute_kernel<<<ropeBlocks, 256, 0, stream>>>(Vp, pos, Vbt, 0, 1.0f, 1);

    dim3 aGrid(CA_N / 64, CA_B * CA_H);                // (32, 32)
    attn_kernel<<<aGrid, 128, 0, stream>>>(Qb, Kb, Vbt, mask, Ob);

    gemm_bf16_kernel<<<gGrid, 256, 0, stream>>>(Ob, projW, out, M, CA_C, CA_C);
}
